// Graphsage_23862838297158
// MI455X (gfx1250) — matrix operand order repaired and checked by reading, not compiled, not run on hardware
//
#include <hip/hip_runtime.h>

#define NN 100000
#define EE 640000
#define FF 128
#define HH 128
#define BN_EPS 1e-5f

typedef __attribute__((ext_vector_type(16))) __bf16 v16bf;
typedef __attribute__((ext_vector_type(8)))  float  v8f;

// ---------------------------------------------------------------- utilities
__global__ void gs_zero_kernel(float* __restrict__ p, int n) {
  int i = blockIdx.x * blockDim.x + threadIdx.x;
  if (i < n) p[i] = 0.0f;
}

__global__ void gs_cvt_bf16_kernel(const float* __restrict__ in,
                                   __bf16* __restrict__ out, int n) {
  int i = blockIdx.x * blockDim.x + threadIdx.x;
  if (i < n) out[i] = (__bf16)in[i];
}

// ------------------------------------------------- edge gather / scatter-add
// One edge per 32-thread group; each thread handles 4 consecutive features.
__global__ void gs_scatter_kernel(const float* __restrict__ x,
                                  const int* __restrict__ src,
                                  const int* __restrict__ dst,
                                  float* __restrict__ agg,
                                  float* __restrict__ cnt) {
  int gid = blockIdx.x * blockDim.x + threadIdx.x;
  int e = gid >> 5;
  if (e >= EE) return;
  int c = gid & 31;
  int s = src[e];
  int d = dst[e];
  const float4 v = *(const float4*)(x + (size_t)s * FF + (size_t)c * 4);
  float* p = agg + (size_t)d * FF + (size_t)c * 4;
  atomicAdd(p + 0, v.x);
  atomicAdd(p + 1, v.y);
  atomicAdd(p + 2, v.z);
  atomicAdd(p + 3, v.w);
  if (c == 0) atomicAdd(cnt + d, 1.0f);
}

// ------------------------------------------------------- fused WMMA GEMM(s)
// Per wave: 16-row x 128-col tile of  Y = (A1*scale) @ Wl^T [+ A2 @ Wr^T] + b
// scale = 1/max(cnt,1) when DUAL (SAGE mean), else 1.
// Also accumulates per-feature sum / sum-of-squares for BatchNorm.
//
// WMMA bf16 lane layouts used here (A/B use contiguous K, not the ISA table order):
//  A 16x32: lane l row M=m0+(l&15); elem e -> K=kk+(l>>4)*16+e (contiguous)
//  B 32x16: lane l col N=n0+(l&15); elem e -> K=kk+(l>>4)*16+e (contiguous)
//  C/D 16x16 f32: lane l col N=n0+(l&15); vgpr v -> M=m0+v+(l>>4)*8
template <bool DUAL>
__global__ void gs_gemm_bn_kernel(const float* __restrict__ A1,
                                  const float* __restrict__ cnt,
                                  const float* __restrict__ A2,
                                  const __bf16* __restrict__ WlB,
                                  const __bf16* __restrict__ WrB,
                                  const float* __restrict__ bias,
                                  float* __restrict__ Y,
                                  float* __restrict__ stat_sum,
                                  float* __restrict__ stat_sq,
                                  int n_tiles) {
  int wave = (blockIdx.x * blockDim.x + threadIdx.x) >> 5;
  int lane = threadIdx.x & 31;
  if (wave >= n_tiles) return;

  const int m0   = wave * 16;
  const int half = lane >> 4;   // 0 or 1
  const int l15  = lane & 15;
  const int rowA = m0 + l15;

  float scale = 1.0f;
  if (DUAL) {
    float c = cnt[rowA];
    scale = 1.0f / fmaxf(c, 1.0f);
  }

  v8f acc[8];
#pragma unroll
  for (int i = 0; i < 8; i++) acc[i] = (v8f){0, 0, 0, 0, 0, 0, 0, 0};

#pragma unroll
  for (int ks = 0; ks < 4; ks++) {
    const int kk = ks * 32;

    v16bf a1, a2;
    const float* pa1 = A1 + (size_t)rowA * HH + kk + half * 16;
#pragma unroll
    for (int e = 0; e < 8; e++) a1[e] = (__bf16)(pa1[e] * scale);
#pragma unroll
    for (int e = 0; e < 8; e++) a1[8 + e] = (__bf16)(pa1[8 + e] * scale);
    if (DUAL) {
      const float* pa2 = A2 + (size_t)rowA * HH + kk + half * 16;
#pragma unroll
      for (int e = 0; e < 8; e++) a2[e] = (__bf16)pa2[e];
#pragma unroll
      for (int e = 0; e < 8; e++) a2[8 + e] = (__bf16)pa2[8 + e];
    }

#pragma unroll
    for (int nt = 0; nt < 8; nt++) {
      const int n = nt * 16 + l15;
      v16bf bl = *(const v16bf*)(WlB + (size_t)n * HH + kk + half * 16);
      acc[nt] = __builtin_amdgcn_wmma_f32_16x16x32_bf16(
          false, a1, false, bl, (short)0, acc[nt], false, false);
      if (DUAL) {
        v16bf br = *(const v16bf*)(WrB + (size_t)n * HH + kk + half * 16);
        acc[nt] = __builtin_amdgcn_wmma_f32_16x16x32_bf16(
            false, a2, false, br, (short)0, acc[nt], false, false);
      }
    }
  }

  // bias, store, BN stat accumulation
#pragma unroll
  for (int nt = 0; nt < 8; nt++) {
    const int n = nt * 16 + l15;
    const float b = bias[n];
    float s = 0.0f, q = 0.0f;
#pragma unroll
    for (int v = 0; v < 8; v++) {
      float val = acc[nt][v] + b;
      int row = m0 + v + half * 8;
      Y[(size_t)row * HH + n] = val;
      s += val;
      q += val * val;
    }
    atomicAdd(&stat_sum[n], s);
    atomicAdd(&stat_sq[n], q);
  }
}

// --------------------------------------------------------------- BN finalize
__global__ void gs_bn_stats_kernel(const float* __restrict__ sum,
                                   const float* __restrict__ sq,
                                   float* __restrict__ mu,
                                   float* __restrict__ rstd, float inv_n) {
  int f = threadIdx.x;  // 128 threads
  float m = sum[f] * inv_n;
  float v = sq[f] * inv_n - m * m;
  mu[f] = m;
  rstd[f] = rsqrtf(fmaxf(v, 0.0f) + BN_EPS);
}

__global__ void gs_bn_relu_kernel(const float* __restrict__ y,
                                  const float* __restrict__ mu,
                                  const float* __restrict__ rstd,
                                  float* __restrict__ xout, int total) {
  int i = blockIdx.x * blockDim.x + threadIdx.x;
  if (i >= total) return;
  int f = i & (HH - 1);
  float v = (y[i] - mu[f]) * rstd[f];
  xout[i] = fmaxf(v, 0.0f);
}

// ------------------------------ fused BN + ReLU + Linear(128 -> 2) epilogue
// One wave per node; lane covers feats lane, lane+32, lane+64, lane+96.
__global__ void gs_final_kernel(const float* __restrict__ y,
                                const float* __restrict__ mu,
                                const float* __restrict__ rstd,
                                const float* __restrict__ W2,
                                const float* __restrict__ b2,
                                float* __restrict__ out, int n_nodes) {
  int wave = (blockIdx.x * blockDim.x + threadIdx.x) >> 5;
  int lane = threadIdx.x & 31;
  if (wave >= n_nodes) return;
  float p0 = 0.0f, p1 = 0.0f;
#pragma unroll
  for (int k = 0; k < 4; k++) {
    int f = lane + k * 32;
    float v = fmaxf((y[(size_t)wave * HH + f] - mu[f]) * rstd[f], 0.0f);
    p0 += v * W2[f];
    p1 += v * W2[HH + f];
  }
#pragma unroll
  for (int off = 16; off > 0; off >>= 1) {
    p0 += __shfl_down(p0, off, 32);
    p1 += __shfl_down(p1, off, 32);
  }
  if (lane == 0) {
    out[(size_t)wave * 2 + 0] = p0 + b2[0];
    out[(size_t)wave * 2 + 1] = p1 + b2[1];
  }
}

// ---------------------------------------------------------------- launcher
extern "C" void kernel_launch(void* const* d_in, const int* in_sizes, int n_in,
                              void* d_out, int out_size, void* d_ws,
                              size_t ws_size, hipStream_t stream) {
  const float* x  = (const float*)d_in[0];
  const int*   ei = (const int*)d_in[1];
  const float* Wl[3] = {(const float*)d_in[2], (const float*)d_in[5], (const float*)d_in[8]};
  const float* bl[3] = {(const float*)d_in[3], (const float*)d_in[6], (const float*)d_in[9]};
  const float* Wr[3] = {(const float*)d_in[4], (const float*)d_in[7], (const float*)d_in[10]};
  const float* W1 = (const float*)d_in[11];
  const float* b1 = (const float*)d_in[12];
  const float* W2 = (const float*)d_in[13];
  const float* b2 = (const float*)d_in[14];
  (void)in_sizes; (void)n_in; (void)out_size; (void)ws_size;

  // workspace layout (floats): agg | cnt | sum | sq | mu | rstd | xbuf | bf16 weights
  float* agg  = (float*)d_ws;
  float* cnt  = agg + (size_t)NN * HH;
  float* sum  = cnt + NN;
  float* sq   = sum + HH;
  float* mu   = sq + HH;
  float* rstd = mu + HH;
  float* xbuf = rstd + HH;
  __bf16* wbf = (__bf16*)(xbuf + (size_t)NN * HH);
  const int WELE = HH * HH;  // 16384

  // 1) convert all weight matrices to bf16 once per launch
  {
    dim3 b(256), g((WELE + 255) / 256);
    for (int i = 0; i < 3; i++) {
      gs_cvt_bf16_kernel<<<g, b, 0, stream>>>(Wl[i], wbf + (size_t)(2 * i) * WELE, WELE);
      gs_cvt_bf16_kernel<<<g, b, 0, stream>>>(Wr[i], wbf + (size_t)(2 * i + 1) * WELE, WELE);
    }
    gs_cvt_bf16_kernel<<<g, b, 0, stream>>>(W1, wbf + (size_t)6 * WELE, WELE);
  }

  const int n_tiles   = NN / 16;                         // 6250 (N divisible by 16)
  const int gemm_grid = (n_tiles + 7) / 8;               // 8 waves / 256-thread block
  const int zero_n    = NN * HH + NN + 4 * HH;           // agg+cnt+stats (contiguous)
  const int elem_n    = NN * HH;

  const float* xcur = x;
  for (int l = 0; l < 3; l++) {
    gs_zero_kernel<<<(zero_n + 255) / 256, 256, 0, stream>>>(agg, zero_n);
    gs_scatter_kernel<<<(EE * 32 + 255) / 256, 256, 0, stream>>>(
        xcur, ei, ei + EE, agg, cnt);
    gs_gemm_bn_kernel<true><<<gemm_grid, 256, 0, stream>>>(
        agg, cnt, xcur, wbf + (size_t)(2 * l) * WELE, wbf + (size_t)(2 * l + 1) * WELE,
        bl[l], agg /* Y aliases agg: safe, wave-private rows */, sum, sq, n_tiles);
    gs_bn_stats_kernel<<<1, HH, 0, stream>>>(sum, sq, mu, rstd, 1.0f / NN);
    gs_bn_relu_kernel<<<(elem_n + 255) / 256, 256, 0, stream>>>(agg, mu, rstd, xbuf, elem_n);
    xcur = xbuf;
  }

  // MLP head: y = x @ W1^T + b1 ; BN ; ReLU ; @ W2^T + b2
  gs_zero_kernel<<<1, 256, 0, stream>>>(sum, 2 * HH);  // just the stats
  gs_gemm_bn_kernel<false><<<gemm_grid, 256, 0, stream>>>(
      xcur, nullptr, nullptr, wbf + (size_t)6 * WELE, nullptr, b1, agg, sum, sq, n_tiles);
  gs_bn_stats_kernel<<<1, HH, 0, stream>>>(sum, sq, mu, rstd, 1.0f / NN);
  gs_final_kernel<<<(NN * 32 + 255) / 256, 256, 0, stream>>>(
      agg, mu, rstd, W2, b2, (float*)d_out, NN);
}
